// RGCModel_48464410968240
// MI455X (gfx1250) — compile-verified, matrix operand order checked by reading, not run on hardware
//
#include <hip/hip_runtime.h>
#include <math.h>

typedef __attribute__((ext_vector_type(16))) __bf16 v16bf;
typedef __attribute__((ext_vector_type(8)))  __bf16 v8bf;
typedef __attribute__((ext_vector_type(8)))  float  v8f;

union V16 { v16bf v; v8bf h[2]; };

#define NNODES 100000
#define NEDGES 1600000
#define FDIM   128
#define CDIM   10
#define GDIM   512
#define BN_EPS 1e-5f
#define NEG_SLOPE 0.2f

// ---------------- precision staging ----------------
__global__ void k_cvt_bf16(const float* __restrict__ in, __bf16* __restrict__ out, int n) {
    int i = blockIdx.x * blockDim.x + threadIdx.x;
    if (i < n) out[i] = (__bf16)in[i];
}
// W[k][n] (row-major KxN) -> Wt[n][k] bf16, so B fragments are contiguous per lane
__global__ void k_cvt_w_t(const float* __restrict__ W, __bf16* __restrict__ wt) {
    int i = blockIdx.x * blockDim.x + threadIdx.x;
    if (i < FDIM * FDIM) {
        int n = i >> 7, k = i & 127;
        wt[i] = (__bf16)W[k * FDIM + n];
    }
}

// ---------------- degree / dinv ----------------
__global__ void k_init_deg(float* deg) {
    int i = blockIdx.x * blockDim.x + threadIdx.x;
    if (i < NNODES) deg[i] = 1.0f;               // self-loop included
}
__global__ void k_count_deg(const int* __restrict__ dst, float* deg) {
    int i = blockIdx.x * blockDim.x + threadIdx.x;
    if (i < NEDGES) atomicAdd(&deg[dst[i]], 1.0f);
}
__global__ void k_rsqrt(float* deg) {
    int i = blockIdx.x * blockDim.x + threadIdx.x;
    if (i < NNODES) deg[i] = rsqrtf(deg[i]);     // becomes dinv
}

// ---------------- WMMA GEMM: out[M x 128] = A[M x 128] @ W[128 x 128] -----
// A pre-converted bf16 row-major; W pre-converted bf16 *transposed* (Wt[n][k]).
// One wave -> one 16x16 output tile. Block = 8 waves = one row-tile x 8 col-tiles.
// Every fragment = two global_load_b128 (16B-aligned); zero VALU conversions.
__global__ void gemm_bf16_wmma(const __bf16* __restrict__ abf,
                               const __bf16* __restrict__ wt,
                               float* __restrict__ out, int M) {
    const int lane = threadIdx.x & 31;
    const int wave = threadIdx.x >> 5;
    const int tm   = blockIdx.x;              // row tile
    const int tn   = wave;                    // col tile (0..7)
    if (tm * 16 >= M) return;                 // wave-uniform: EXEC stays all-1s

    const int mrow  = tm * 16 + (lane & 15);
    const int ncol  = tn * 16 + (lane & 15);
    const int khalf = (lane & 16) ? 8 : 0;    // ISA 16-bit A/B lane layout

    const __bf16* arow = abf + (size_t)mrow * FDIM;
    const __bf16* brow = wt  + (size_t)ncol * FDIM;

    v8f c = {};
    #pragma unroll
    for (int kt = 0; kt < 4; ++kt) {          // K = 128 in steps of 32
        const int kb = kt * 32 + khalf;
        V16 a, b;
        a.h[0] = *(const v8bf*)(arow + kb);
        a.h[1] = *(const v8bf*)(arow + kb + 16);
        b.h[0] = *(const v8bf*)(brow + kb);
        b.h[1] = *(const v8bf*)(brow + kb + 16);
        c = __builtin_amdgcn_wmma_f32_16x16x32_bf16(
                false, a.v, false, b.v, (short)0, c, false, false);
    }

    const int mo = (lane & 16) ? 8 : 0;       // C/D layout: M = r + mo
    #pragma unroll
    for (int r = 0; r < 8; ++r)
        out[(size_t)(tm * 16 + r + mo) * FDIM + ncol] = c[r];
}

// ---------------- edge aggregation: agg[dst] += norm * h[src] -------------
// One wave per edge; lane handles 4 contiguous features (float4).
__global__ void k_aggregate(const float* __restrict__ h,
                            const int* __restrict__ src,
                            const int* __restrict__ dst,
                            const float* __restrict__ dinv,
                            float* __restrict__ agg) {
    int e    = (blockIdx.x * blockDim.x + threadIdx.x) >> 5;
    int lane = threadIdx.x & 31;
    if (e >= NEDGES) return;
    int s = src[e], d = dst[e];
    float norm = dinv[s] * dinv[d];
    const float4* hs = (const float4*)(h + (size_t)s * FDIM);
    float4 v = hs[lane];
    float* ad = agg + (size_t)d * FDIM + lane * 4;
    atomicAdd(ad + 0, norm * v.x);
    atomicAdd(ad + 1, norm * v.y);
    atomicAdd(ad + 2, norm * v.z);
    atomicAdd(ad + 3, norm * v.w);
}

// ---------------- combine: leaky(agg + dinv^2 * h + b) --------------------
// bf16 output variant feeds the next WMMA layer directly (GEMM would truncate
// to bf16 anyway, so no extra precision loss).
__global__ void k_combine_bf16(const float* __restrict__ gemm,
                               const float* __restrict__ agg,
                               const float* __restrict__ dinv,
                               const float* __restrict__ bias,
                               __bf16* __restrict__ out) {
    int t = blockIdx.x * blockDim.x + threadIdx.x;
    if (t >= NNODES * FDIM) return;
    int row = t >> 7, col = t & 127;
    float di = dinv[row];
    float v  = agg[t] + di * di * gemm[t] + bias[col];
    out[t]   = (__bf16)(v > 0.0f ? v : NEG_SLOPE * v);
}
__global__ void k_combine_f32(const float* gemm, const float* agg,
                              const float* __restrict__ dinv,
                              const float* __restrict__ bias,
                              float* out) {                  // may alias (elementwise)
    int t = blockIdx.x * blockDim.x + threadIdx.x;
    if (t >= NNODES * FDIM) return;
    int row = t >> 7, col = t & 127;
    float di = dinv[row];
    float v  = agg[t] + di * di * gemm[t] + bias[col];
    out[t]   = v > 0.0f ? v : NEG_SLOPE * v;
}

// ---------------- BN stats: per-channel sum / sumsq -----------------------
__global__ void k_bnstats(const float* __restrict__ h, float* __restrict__ stats) {
    int col = threadIdx.x;                     // 128 threads
    int r0  = blockIdx.x * 256;
    float s = 0.0f, s2 = 0.0f;
    for (int r = 0; r < 256; ++r) {
        int row = r0 + r;
        if (row < NNODES) {
            float v = h[(size_t)row * FDIM + col];
            s += v; s2 += v * v;
        }
    }
    atomicAdd(&stats[col], s);
    atomicAdd(&stats[FDIM + col], s2);
}

// ---------------- BN apply + global mean-pool (sums) ----------------------
__global__ void k_bnpool(const float* __restrict__ h,
                         const float* __restrict__ stats,
                         const float* __restrict__ gamma,
                         const float* __restrict__ beta,
                         const int* __restrict__ batch,
                         float* __restrict__ pooled) {
    int t = blockIdx.x * blockDim.x + threadIdx.x;
    if (t >= NNODES * FDIM) return;
    int row = t >> 7, col = t & 127;
    const float invN = 1.0f / (float)NNODES;
    float mu  = stats[col] * invN;
    float var = stats[FDIM + col] * invN - mu * mu;   // biased, as torch
    float y   = (h[t] - mu) * rsqrtf(var + BN_EPS) * gamma[col] + beta[col];
    atomicAdd(&pooled[(size_t)batch[row] * FDIM + col], y);
}
__global__ void k_count(const int* __restrict__ batch, float* __restrict__ cnt) {
    int i = blockIdx.x * blockDim.x + threadIdx.x;
    if (i < NNODES) atomicAdd(&cnt[batch[i]], 1.0f);
}

// ---------------- classifier + log_softmax --------------------------------
__global__ void k_classify(const float* __restrict__ pooled,
                           const float* __restrict__ cnt,
                           const float* __restrict__ fcW,   // [H, C]
                           const float* __restrict__ fcb,   // [C]
                           float* __restrict__ out) {       // [G, C]
    __shared__ float sd[FDIM * CDIM];
    int g = blockIdx.x, t = threadIdx.x;       // 128 threads
    float v = pooled[(size_t)g * FDIM + t] / fmaxf(cnt[g], 1.0f);
    for (int c = 0; c < CDIM; ++c) sd[t * CDIM + c] = v * fcW[t * CDIM + c];
    __syncthreads();
    for (int off = 64; off >= 1; off >>= 1) {
        if (t < off)
            for (int c = 0; c < CDIM; ++c)
                sd[t * CDIM + c] += sd[(t + off) * CDIM + c];
        __syncthreads();
    }
    if (t == 0) {
        float logits[CDIM], m = -1e30f;
        for (int c = 0; c < CDIM; ++c) { logits[c] = sd[c] + fcb[c]; m = fmaxf(m, logits[c]); }
        float se = 0.0f;
        for (int c = 0; c < CDIM; ++c) se += __expf(logits[c] - m);
        float lse = m + __logf(se);
        for (int c = 0; c < CDIM; ++c) out[g * CDIM + c] = logits[c] - lse;
    }
}

extern "C" void kernel_launch(void* const* d_in, const int* in_sizes, int n_in,
                              void* d_out, int out_size, void* d_ws, size_t ws_size,
                              hipStream_t stream) {
    const float* x     = (const float*)d_in[0];
    const int*   ei    = (const int*)  d_in[1];
    const int*   srcI  = ei;
    const int*   dstI  = ei + NEDGES;
    const int*   batch = (const int*)  d_in[2];
    const float* W1 = (const float*)d_in[3];  const float* b1 = (const float*)d_in[4];
    const float* W2 = (const float*)d_in[5];  const float* b2 = (const float*)d_in[6];
    const float* gamma = (const float*)d_in[7];
    const float* beta  = (const float*)d_in[8];
    const float* fcW = (const float*)d_in[9]; const float* fcb = (const float*)d_in[10];
    float* out = (float*)d_out;

    char*  ws = (char*)d_ws;
    size_t nh  = (size_t)NNODES * FDIM * sizeof(float);   // 51.2 MB
    float*  buf0   = (float*)ws;                          // gemm-out / act2
    float*  buf1   = (float*)(ws + nh);                   // agg
    __bf16* xbf    = (__bf16*)(ws + 2 * nh);              // bf16 activations (reused)
    __bf16* w1t    = xbf + (size_t)NNODES * FDIM;         // 128x128 bf16
    __bf16* w2t    = w1t + FDIM * FDIM;
    float*  dinv   = (float*)(w2t + FDIM * FDIM);         // N
    float*  stats  = dinv + NNODES;                       // 2*H
    float*  pooled = stats + 2 * FDIM;                    // G*H
    float*  cnt    = pooled + (size_t)GDIM * FDIM;        // G

    const int NHT = NNODES * FDIM;

    // stage weights (transposed) and input features as bf16
    k_cvt_w_t <<<(FDIM * FDIM + 255) / 256, 256, 0, stream>>>(W1, w1t);
    k_cvt_w_t <<<(FDIM * FDIM + 255) / 256, 256, 0, stream>>>(W2, w2t);
    k_cvt_bf16<<<(NHT + 255) / 256, 256, 0, stream>>>(x, xbf, NHT);

    // degree -> dinv
    k_init_deg <<<(NNODES + 255) / 256, 256, 0, stream>>>(dinv);
    k_count_deg<<<(NEDGES + 255) / 256, 256, 0, stream>>>(dstI, dinv);
    k_rsqrt    <<<(NNODES + 255) / 256, 256, 0, stream>>>(dinv);

    // ---- GCN layer 1 ----
    gemm_bf16_wmma<<<NNODES / 16, 256, 0, stream>>>(xbf, w1t, buf0, NNODES);
    hipMemsetAsync(buf1, 0, nh, stream);
    k_aggregate   <<<NEDGES / 8, 256, 0, stream>>>(buf0, srcI, dstI, dinv, buf1);
    k_combine_bf16<<<(NHT + 255) / 256, 256, 0, stream>>>(buf0, buf1, dinv, b1, xbf);

    // ---- GCN layer 2 ----
    gemm_bf16_wmma<<<NNODES / 16, 256, 0, stream>>>(xbf, w2t, buf0, NNODES);
    hipMemsetAsync(buf1, 0, nh, stream);
    k_aggregate  <<<NEDGES / 8, 256, 0, stream>>>(buf0, srcI, dstI, dinv, buf1);
    k_combine_f32<<<(NHT + 255) / 256, 256, 0, stream>>>(buf0, buf1, dinv, b2, buf0);

    // ---- BN + pool + classify ----
    hipMemsetAsync(stats, 0, (2 * FDIM + (size_t)GDIM * FDIM + GDIM) * sizeof(float), stream);
    k_bnstats<<<(NNODES + 255) / 256, 128, 0, stream>>>(buf0, stats);
    k_bnpool <<<(NHT + 255) / 256, 256, 0, stream>>>(buf0, stats, gamma, beta, batch, pooled);
    k_count  <<<(NNODES + 255) / 256, 256, 0, stream>>>(batch, cnt);
    k_classify<<<GDIM, 128, 0, stream>>>(pooled, cnt, fcW, fcb, out);
}
